// MTLWithGraph_57569741636288
// MI455X (gfx1250) — compile-verified, matrix operand order checked by reading, not run on hardware
//
#include <hip/hip_runtime.h>
#include <stdint.h>

// ---------------- problem constants ----------------
#define B_TOTAL   65536
#define IN_DIM    512
#define H0_DIM    256
#define H1_DIM    128
#define HEAD_H    64
#define N_DIS     12
#define KMAXP     3
#define BT        64      // batch rows per workgroup
#define KC        128     // K-chunk for GEMM1
#define NTHREADS  256     // 8 wave32
#define EPSV      1e-5f

// ---------------- padded LDS row strides (in halves) ----------------
// Rows padded by +16 halves (32 B) to break power-of-two bank patterns while
// keeping every v16h / ds_load_b128 access 32-byte aligned.
#define XS   144    // bufX row stride (row length KC=128)
#define HS   272    // bufH row stride (row length H0=256)
#define ZS   144    // bufZ row stride (row length H1=128)
#define WS1  144    // bufW stride, phase 1 (row length KC=128)
#define WS2  272    // bufW stride, phase 2 (row length H0=256)
#define WS4  144    // bufW stride, phase 4 (row length H1=128)

// ---------------- LDS layout (bytes) ----------------
#define OFF_W    0                      // 73728 B : transposed f16 weight panel (padded)
#define OFF_X    73728                  // 18432 B : f16 x chunk [BT][XS]
#define OFF_H    92160                  // 34816 B : f16 hidden  [BT][HS]
#define OFF_XF   92160                  // 32768 B : f32 TDM staging (aliases bufH)
#define OFF_Z    126976                 // 18432 B : f16 z       [BT][ZS]
#define OFF_P    145408                 //  3072 B : f32 pass-1 preds [BT][N_DIS]
#define OFF_ACC  148480                 //   256 B : f32 head reduction [BT]
#define SMEM_BYTES 148736

typedef __attribute__((ext_vector_type(16))) _Float16     v16h;
typedef __attribute__((ext_vector_type(8)))  float        v8f;
typedef __attribute__((ext_vector_type(4)))  unsigned int u32x4;
typedef __attribute__((ext_vector_type(8)))  int          i32x8;
typedef __attribute__((ext_vector_type(4)))  int          i32x4;

#if __has_builtin(__builtin_amdgcn_tensor_load_to_lds) && __has_builtin(__builtin_amdgcn_s_wait_tensorcnt)
#define HAVE_TDM 1
#else
#define HAVE_TDM 0
#endif

// Scheduler fence: keep the fragment-load group and the WMMA group separate so
// all ds_loads of a K-step issue before the first WMMA (one wait, then
// back-to-back WMMA issue; next step's loads may still sink under the WMMAs).
#if __has_builtin(__builtin_amdgcn_sched_barrier)
#define SCHED_FENCE() __builtin_amdgcn_sched_barrier(0)
#else
#define SCHED_FENCE() ((void)0)
#endif

#if HAVE_TDM
// 2D TDM tile load: global f32 tile (tile_h rows x tile_w elems, row stride
// row_stride elems) -> contiguous LDS at byte offset lds_byte_off.
// D# packing per CDNA5 ISA §8.3/§8.4. Groups 2/3 zero (2D tensor).
__device__ __forceinline__ void tdm_load_tile_f32(
    unsigned lds_byte_off, const float* gsrc,
    unsigned tile_w, unsigned tile_h,
    unsigned tensor_w, unsigned tensor_h, unsigned row_stride)
{
  unsigned long long ga = (unsigned long long)(uintptr_t)gsrc;
  u32x4 g0;
  g0.x = 1u;                                                // count=1 (valid), user mode
  g0.y = lds_byte_off;                                      // lds_addr
  g0.z = (unsigned)ga;                                      // global_addr[31:0]
  g0.w = (unsigned)((ga >> 32) & 0x01FFFFFFu) | (2u << 30); // global_addr[56:32] | type=2
  i32x8 g1;
  g1[0] = (int)(2u << 16);                                  // data_size=2 (4 bytes)
  g1[1] = (int)((tensor_w & 0xFFFFu) << 16);                // tensor_dim0[15:0]
  g1[2] = (int)((tensor_w >> 16) | ((tensor_h & 0xFFFFu) << 16)); // dim0 hi | dim1 lo
  g1[3] = (int)((tensor_h >> 16) | ((tile_w & 0xFFFFu) << 16));   // dim1 hi | tile_dim0
  g1[4] = (int)(tile_h & 0xFFFFu);                          // tile_dim1, tile_dim2=0
  g1[5] = (int)row_stride;                                  // tensor_dim0_stride[31:0]
  g1[6] = 0;
  g1[7] = 0;
  i32x4 gz = {0, 0, 0, 0};
#if defined(__clang_major__) && (__clang_major__ >= 23)
  i32x8 gz8 = {0, 0, 0, 0, 0, 0, 0, 0};
  __builtin_amdgcn_tensor_load_to_lds(g0, g1, gz, gz, gz8, 0);
#else
  __builtin_amdgcn_tensor_load_to_lds(g0, g1, gz, gz, 0);
#endif
}
#endif

__global__ __launch_bounds__(NTHREADS) void fused_dag_mtl(
    const float* __restrict__ x,
    const float* __restrict__ encW0, const float* __restrict__ encB0,
    const float* __restrict__ bn0g,  const float* __restrict__ bn0b,
    const float* __restrict__ bn0m,  const float* __restrict__ bn0v,
    const float* __restrict__ encW1, const float* __restrict__ encB1,
    const float* __restrict__ bn1g,  const float* __restrict__ bn1b,
    const float* __restrict__ bn1m,  const float* __restrict__ bn1v,
    const float* __restrict__ wInit, const float* __restrict__ bInit,
    const float* __restrict__ Wz,    const float* __restrict__ Wp,
    const float* __restrict__ bhp,   const float* __restrict__ Wo,
    const float* __restrict__ bo,
    const float* __restrict__ predW, const int* __restrict__ predIdx,
    float* __restrict__ out)
{
  extern __shared__ char smem[];
  _Float16* bufW = (_Float16*)(smem + OFF_W);
  _Float16* bufX = (_Float16*)(smem + OFF_X);
  _Float16* bufH = (_Float16*)(smem + OFF_H);
  _Float16* bufZ = (_Float16*)(smem + OFF_Z);
  float*    bufF = (float*)(smem + OFF_XF);   // aliases bufH (disjoint lifetime)
  float*    bufP = (float*)(smem + OFF_P);
  float*    bufA = (float*)(smem + OFF_ACC);

  const int tid  = (int)threadIdx.x;
  const int wave = tid >> 5;
  const int lane = tid & 31;
  const int lo16 = lane & 15;
  const int hiK  = (lane >> 4) << 4;   // +16 halves for upper half-wave (K split)
  const int hiM  = (lane >> 4) << 3;   // +8 rows for upper half-wave (C/D layout)
  const long long row0 = (long long)blockIdx.x * BT;

  // =============== Phase 1: h = ReLU(BN0(x @ W0 + b0)) -> bufH (f16) ===============
  {
    const int rt = wave & 3;      // 4 row tiles of 16
    const int cg = wave >> 2;     // 2 col groups x 8 tiles of 16 (H0=256)
    v8f acc[8] = {};
    for (int kc = 0; kc < IN_DIM; kc += KC) {
      // stage x chunk (f32) -> bufF
#if HAVE_TDM
      if (wave == 0) {
        tdm_load_tile_f32(OFF_XF, x + row0 * IN_DIM + kc,
                          KC, BT, IN_DIM, B_TOTAL, IN_DIM);
        __builtin_amdgcn_s_wait_tensorcnt(0);
      }
#else
      for (int i = tid; i < BT * KC; i += NTHREADS) {
        int r = i >> 7, c = i & (KC - 1);
        bufF[i] = x[(row0 + r) * IN_DIM + kc + c];
      }
#endif
      // pull the next W0 chunk toward the caches while this one computes
      if (kc + KC < IN_DIM)
        __builtin_prefetch(encW0 + (long long)(kc + KC) * H0_DIM + tid * 16, 0, 0);
      // stage W0 chunk transposed: bufW[n*WS1 + k] = W0[(kc+k)*H0 + n]
      for (int i = tid; i < KC * H0_DIM; i += NTHREADS) {
        int k = i >> 8;            // / H0_DIM
        int n = i & (H0_DIM - 1);
        bufW[n * WS1 + k] = (_Float16)encW0[(long long)(kc + k) * H0_DIM + n];
      }
      __syncthreads();
      // convert x chunk to f16 (contiguous in, padded-row out)
      for (int i = tid; i < BT * KC; i += NTHREADS) {
        int r = i >> 7, c = i & (KC - 1);
        bufX[r * XS + c] = (_Float16)bufF[i];
      }
      __syncthreads();
      // WMMA over this K chunk: batch all fragment loads, then WMMA back-to-back
      const int m = 16 * rt + lo16;
      #pragma unroll
      for (int kk = 0; kk < KC; kk += 32) {
        v16h bf[8];
        #pragma unroll
        for (int t = 0; t < 8; ++t) {
          const int n = cg * 128 + t * 16 + lo16;
          bf[t] = *(const v16h*)(bufW + n * WS1 + kk + hiK);
        }
        const v16h a = *(const v16h*)(bufX + m * XS + kk + hiK);
        SCHED_FENCE();
        #pragma unroll
        for (int t = 0; t < 8; ++t)
          acc[t] = __builtin_amdgcn_wmma_f32_16x16x32_f16(
              false, a, false, bf[t], (short)0, acc[t], false, false);
      }
      __syncthreads();             // before restaging bufX/bufW/bufF
    }
    // epilogue: bias + BN0 + ReLU -> bufH f16 (overwrites dead bufF region)
    #pragma unroll
    for (int t = 0; t < 8; ++t) {
      const int n  = cg * 128 + t * 16 + lo16;
      const float sc = bn0g[n] * __frsqrt_rn(bn0v[n] + EPSV);
      const float sh = bn0b[n] + (encB0[n] - bn0m[n]) * sc;
      #pragma unroll
      for (int r = 0; r < 8; ++r) {
        const int mm = 16 * rt + r + hiM;
        float v = fmaf(acc[t][r], sc, sh);
        bufH[mm * HS + n] = (_Float16)fmaxf(v, 0.0f);
      }
    }
  }
  __syncthreads();

  // =============== Phase 2: z = ReLU(BN1(h @ W1 + b1)) -> bufZ (f16) ===============
  {
    // stage W1 transposed: bufW[n*WS2 + k] = W1[k*H1 + n]
    for (int i = tid; i < H0_DIM * H1_DIM; i += NTHREADS) {
      int k = i >> 7;              // / H1_DIM
      int n = i & (H1_DIM - 1);
      bufW[n * WS2 + k] = (_Float16)encW1[k * H1_DIM + n];
    }
    __syncthreads();
    const int rt = wave & 3;
    const int cg = wave >> 2;      // 2 col groups x 4 tiles (H1=128)
    v8f acc[4] = {};
    const int m = 16 * rt + lo16;
    #pragma unroll
    for (int kk = 0; kk < H0_DIM; kk += 32) {
      v16h bf[4];
      #pragma unroll
      for (int t = 0; t < 4; ++t) {
        const int n = cg * 64 + t * 16 + lo16;
        bf[t] = *(const v16h*)(bufW + n * WS2 + kk + hiK);
      }
      const v16h a = *(const v16h*)(bufH + m * HS + kk + hiK);
      SCHED_FENCE();
      #pragma unroll
      for (int t = 0; t < 4; ++t)
        acc[t] = __builtin_amdgcn_wmma_f32_16x16x32_f16(
            false, a, false, bf[t], (short)0, acc[t], false, false);
    }
    #pragma unroll
    for (int t = 0; t < 4; ++t) {
      const int n  = cg * 64 + t * 16 + lo16;
      const float sc = bn1g[n] * __frsqrt_rn(bn1v[n] + EPSV);
      const float sh = bn1b[n] + (encB1[n] - bn1m[n]) * sc;
      #pragma unroll
      for (int r = 0; r < 8; ++r) {
        const int mm = 16 * rt + r + hiM;
        float v = fmaf(acc[t][r], sc, sh);
        bufZ[mm * ZS + n] = (_Float16)fmaxf(v, 0.0f);
      }
    }
  }
  __syncthreads();

  // =============== Phase 3: p = sigmoid(z @ w_init^T + b_init) -> bufP ===============
  for (int i = tid; i < BT * N_DIS; i += NTHREADS) {
    const int m = i / N_DIS;
    const int d = i - m * N_DIS;
    float s = bInit[d];
    #pragma unroll 8
    for (int k = 0; k < H1_DIM; ++k)
      s += (float)bufZ[m * ZS + k] * wInit[d * H1_DIM + k];
    bufP[i] = 1.0f / (1.0f + __expf(-s));
  }
  __syncthreads();

  // =============== Phase 4: per-disease heads ===============
  {
    const int rt = wave & 3;
    const int cg = wave >> 2;      // 2 groups x 2 col tiles (HEAD_H=64)
    const int m  = 16 * rt + lo16;
    for (int d = 0; d < N_DIS; ++d) {
#if HAVE_TDM
      // stage Wz[d] (H1 x HEAD_H f32) via TDM into bufF, then transpose+convert
      if (wave == 0) {
        tdm_load_tile_f32(OFF_XF, Wz + (long long)d * H1_DIM * HEAD_H,
                          HEAD_H, H1_DIM, HEAD_H, (unsigned)(N_DIS * H1_DIM), HEAD_H);
        __builtin_amdgcn_s_wait_tensorcnt(0);
      }
      if (tid < BT) bufA[tid] = 0.0f;
      __syncthreads();
      for (int i = tid; i < H1_DIM * HEAD_H; i += NTHREADS) {
        int k = i >> 6;            // / HEAD_H
        int n = i & (HEAD_H - 1);
        bufW[n * WS4 + k] = (_Float16)bufF[i];
      }
#else
      if (tid < BT) bufA[tid] = 0.0f;
      for (int i = tid; i < H1_DIM * HEAD_H; i += NTHREADS) {
        int k = i >> 6;
        int n = i & (HEAD_H - 1);
        bufW[n * WS4 + k] = (_Float16)Wz[((long long)d * H1_DIM + k) * HEAD_H + n];
      }
#endif
      __syncthreads();
      v8f acc[2] = {};
      #pragma unroll
      for (int kk = 0; kk < H1_DIM; kk += 32) {
        v16h bf[2];
        #pragma unroll
        for (int t = 0; t < 2; ++t) {
          const int n = (cg * 2 + t) * 16 + lo16;
          bf[t] = *(const v16h*)(bufW + n * WS4 + kk + hiK);
        }
        const v16h a = *(const v16h*)(bufZ + m * ZS + kk + hiK);
        SCHED_FENCE();
        #pragma unroll
        for (int t = 0; t < 2; ++t)
          acc[t] = __builtin_amdgcn_wmma_f32_16x16x32_f16(
              false, a, false, bf[t], (short)0, acc[t], false, false);
      }
      // epilogue: + bh + DAG predecessor term, ReLU, dot with Wo via LDS atomics
      #pragma unroll
      for (int t = 0; t < 2; ++t) {
        const int n = (cg * 2 + t) * 16 + lo16;
        const float bhn = bhp[d * HEAD_H + n];
        const float won = Wo[d * HEAD_H + n];
        float wpn[KMAXP];
        int   pix[KMAXP];
        #pragma unroll
        for (int j = 0; j < KMAXP; ++j) {
          wpn[j] = Wp[((long long)d * KMAXP + j) * HEAD_H + n] * predW[d * KMAXP + j];
          pix[j] = predIdx[d * KMAXP + j];
        }
        #pragma unroll
        for (int r = 0; r < 8; ++r) {
          const int mm = 16 * rt + r + hiM;
          float s = acc[t][r] + bhn;
          #pragma unroll
          for (int j = 0; j < KMAXP; ++j)
            s += bufP[mm * N_DIS + pix[j]] * wpn[j];
          s = fmaxf(s, 0.0f);
          atomicAdd(&bufA[mm], s * won);    // ds_add_f32
        }
      }
      __syncthreads();
      if (tid < BT) {
        const float o = 1.0f / (1.0f + __expf(-(bufA[tid] + bo[d])));
        out[(row0 + tid) * N_DIS + d] = o;
      }
      __syncthreads();               // before next d restages bufW/bufF/bufA
    }
  }
}

extern "C" void kernel_launch(void* const* d_in, const int* in_sizes, int n_in,
                              void* d_out, int out_size, void* d_ws, size_t ws_size,
                              hipStream_t stream) {
  (void)in_sizes; (void)n_in; (void)out_size; (void)d_ws; (void)ws_size;
  const float* x     = (const float*)d_in[0];
  const float* eW0   = (const float*)d_in[1];
  const float* eB0   = (const float*)d_in[2];
  const float* g0    = (const float*)d_in[3];
  const float* b0    = (const float*)d_in[4];
  const float* m0    = (const float*)d_in[5];
  const float* v0    = (const float*)d_in[6];
  const float* eW1   = (const float*)d_in[7];
  const float* eB1   = (const float*)d_in[8];
  const float* g1    = (const float*)d_in[9];
  const float* b1    = (const float*)d_in[10];
  const float* m1    = (const float*)d_in[11];
  const float* v1    = (const float*)d_in[12];
  const float* wI    = (const float*)d_in[13];
  const float* bI    = (const float*)d_in[14];
  const float* Wz    = (const float*)d_in[15];
  const float* Wp    = (const float*)d_in[16];
  const float* bh    = (const float*)d_in[17];
  const float* Wo    = (const float*)d_in[18];
  const float* bo    = (const float*)d_in[19];
  const float* pw    = (const float*)d_in[20];
  const int*   pidx  = (const int*)d_in[21];
  float* out = (float*)d_out;

  (void)hipFuncSetAttribute(reinterpret_cast<const void*>(fused_dag_mtl),
                            hipFuncAttributeMaxDynamicSharedMemorySize, SMEM_BYTES);
  dim3 grid(B_TOTAL / BT);
  fused_dag_mtl<<<grid, NTHREADS, SMEM_BYTES, stream>>>(
      x, eW0, eB0, g0, b0, m0, v0, eW1, eB1, g1, b1, m1, v1,
      wI, bI, Wz, Wp, bh, Wo, bo, pw, pidx, out);
}
